// CandidateRetrieval_16681652977973
// MI455X (gfx1250) — compile-verified, matrix operand order checked by reading, not run on hardware
//
#include <hip/hip_runtime.h>
#include <hip/hip_bf16.h>

typedef __attribute__((ext_vector_type(16))) _Float16 v16h;
typedef __attribute__((ext_vector_type(8)))  _Float16 v8h;
typedef __attribute__((ext_vector_type(4)))  _Float16 v4h;
typedef __attribute__((ext_vector_type(8)))  float    v8f;
typedef __attribute__((ext_vector_type(4)))  unsigned int v4u;
typedef __attribute__((ext_vector_type(8)))  int      v8i;
typedef __attribute__((ext_vector_type(4)))  int      v4i;

#define DD 512      // feature dim
#define KK 64       // top-k
#define MT 32       // query rows per workgroup
#define NT 64       // embedding cols per tile
#define KH 256      // K elements staged per TDM stage (half of DD)

#if __has_builtin(__builtin_amdgcn_tensor_load_to_lds)
#define USE_TDM 1
#if __has_include(<hip/amd_detail/amd_gfx1250_TDM.h>)
#define TDM_6ARG 1   // therock-10.0 headers -> clang-23 6-arg builtin
#endif
#endif

// ---------------------------------------------------------------------------
// Kernel 1: L2-normalize rows of both matrices, convert to f16 in workspace.
// ---------------------------------------------------------------------------
__global__ void cr_normalize_f16(const float* __restrict__ z,
                                 const float* __restrict__ e,
                                 _Float16* __restrict__ qh,
                                 _Float16* __restrict__ eh,
                                 int B, int N) {
    const int lane = threadIdx.x & 31;
    const int wave = threadIdx.x >> 5;
    const int row  = blockIdx.x * 8 + wave;
    if (row >= B + N) return;

    const float* src;
    _Float16*    dst;
    int r;
    if (row < B) { src = z; dst = qh; r = row; }
    else         { src = e; dst = eh; r = row - B; }

    const float4* p = (const float4*)(src + (size_t)r * DD);
    float4 v0 = p[lane +  0];
    float4 v1 = p[lane + 32];
    float4 v2 = p[lane + 64];
    float4 v3 = p[lane + 96];

    float ss = v0.x*v0.x + v0.y*v0.y + v0.z*v0.z + v0.w*v0.w
             + v1.x*v1.x + v1.y*v1.y + v1.z*v1.z + v1.w*v1.w
             + v2.x*v2.x + v2.y*v2.y + v2.z*v2.z + v2.w*v2.w
             + v3.x*v3.x + v3.y*v3.y + v3.z*v3.z + v3.w*v3.w;
    #pragma unroll
    for (int off = 16; off > 0; off >>= 1)
        ss += __shfl_xor(ss, off, 32);

    const float scale = 1.0f / fmaxf(sqrtf(ss), 1e-12f);

    _Float16* drow = dst + (size_t)r * DD;
    float4 vv[4] = {v0, v1, v2, v3};
    #pragma unroll
    for (int c = 0; c < 4; ++c) {
        v4h h;
        h.x = (_Float16)(vv[c].x * scale);
        h.y = (_Float16)(vv[c].y * scale);
        h.z = (_Float16)(vv[c].z * scale);
        h.w = (_Float16)(vv[c].w * scale);
        *(v4h*)(drow + (lane + c * 32) * 4) = h;
    }
}

// ---------------------------------------------------------------------------
// TDM issue: load a 64-row x 256-elem f16 tile (row stride 512) into LDS.
// D# per CDNA5 ISA ch.8: group0 = {count/flags, lds_addr, global_addr, type=2},
// group1 = dims/strides (data_size=1 -> 2 bytes).
// ---------------------------------------------------------------------------
#ifdef USE_TDM
__device__ __forceinline__ void tdm_issue_tile(const _Float16* gptr,
                                               unsigned int ldsAddr) {
    unsigned long long ga = (unsigned long long)(uintptr_t)gptr;
    v4u g0;
    g0.x = 1u;                                          // count=1, user mode
    g0.y = ldsAddr;                                     // lds_addr [63:32]
    g0.z = (unsigned int)(ga & 0xFFFFFFFFu);            // global_addr lo
    g0.w = (unsigned int)((ga >> 32) & 0x01FFFFFFu)     // global_addr hi (57b)
         | (2u << 30);                                  // type=2 ("image")
    v8i g1;
    g1[0] = (int)(1u << 16);        // workgroup_mask=0, data_size=1 (2B)
    g1[1] = (int)(512u << 16);      // tensor_dim0[15:0]=512 @ bits[63:48]
    g1[2] = (int)(64u  << 16);      // tensor_dim1[15:0]=64  @ bits[111:80]
    g1[3] = (int)(256u << 16);      // tile_dim0=256 @ bits[127:112]
    g1[4] = 64;                     // tile_dim1=64  @ bits[143:128]
    g1[5] = 512;                    // tensor_dim0_stride=512 @ bits[207:160]
    g1[6] = 0;
    g1[7] = 0;
    v4i z4 = {0, 0, 0, 0};
#ifdef TDM_6ARG
    v8i z8 = {0, 0, 0, 0, 0, 0, 0, 0};
    __builtin_amdgcn_tensor_load_to_lds(g0, g1, z4, z4, z8, 0);
#else
    __builtin_amdgcn_tensor_load_to_lds(g0, g1, z4, z4, 0);
#endif
}
#endif

// One WMMA sweep over a 256-deep K stage from an LDS column.
template <int AOFF>
__device__ __forceinline__ void compute_stage(v8f& c, const v16h* aF,
                                              const _Float16* bcol, int kSel) {
    #pragma unroll
    for (int ks = 0; ks < KH / 32; ++ks) {
        v8h blo = *(const v8h*)(bcol + ks * 32 + kSel);
        v8h bhi = *(const v8h*)(bcol + ks * 32 + kSel + 16);
        v16h b = __builtin_shufflevector(blo, bhi,
                   0,1,2,3,4,5,6,7,8,9,10,11,12,13,14,15);
        c = __builtin_amdgcn_wmma_f32_16x16x32_f16(
                false, aF[AOFF + ks], false, b, (short)0, c, false, false);
    }
}

// ---------------------------------------------------------------------------
// Kernel 2: TDM-staged WMMA GEMM (Q_h x E_h^T) + streaming per-row top-64.
// 8 waves (2x4 tile grid) over a 32x64 output tile; B double-buffered in LDS.
// ---------------------------------------------------------------------------
__global__ void __launch_bounds__(256, 2)
cr_gemm_topk(const _Float16* __restrict__ qh,
             const _Float16* __restrict__ eh,
             float* __restrict__ simsOut,
             int*   __restrict__ idxOut,
             int B, int N) {
    __shared__ _Float16 sB[2][NT][KH];    // 2 x 32KB ping-pong B tiles
    __shared__ float sTile[MT][NT + 1];   // score tile (padded)
    __shared__ float sTopV[MT][KK];
    __shared__ int   sTopI[MT][KK];
    __shared__ float sMin[MT];
    __shared__ int   sMinPos[MT];

    const int tid  = threadIdx.x;
    const int lane = tid & 31;
    const int wave = tid >> 5;
    const int mi   = wave & 1;            // row tile
    const int ni   = wave >> 1;           // col tile
    const int mRowBlk = blockIdx.x * MT;

    for (int i = tid; i < MT * KK; i += 256) {
        ((float*)sTopV)[i] = -__builtin_inff();
        ((int*)sTopI)[i]   = 0;
    }
    if (tid < MT) { sMin[tid] = -__builtin_inff(); sMinPos[tid] = 0; }

    // Preload all A fragments (16 rows x K=512) into registers:
    // lane 0-15 -> row m, K {0..7,16..23}+32k; lane 16-31 -> K {8..15,24..31}.
    const int  mRow = mRowBlk + mi * 16 + (lane & 15);
    const int  kSel = (lane & 16) ? 8 : 0;
    const _Float16* aBase = qh + (size_t)mRow * DD;
    v16h aF[16];
    #pragma unroll
    for (int kk = 0; kk < 16; ++kk) {
        v8h lo = *(const v8h*)(aBase + kk * 32 + kSel);
        v8h hi = *(const v8h*)(aBase + kk * 32 + kSel + 16);
        aF[kk] = __builtin_shufflevector(lo, hi,
                   0,1,2,3,4,5,6,7,8,9,10,11,12,13,14,15);
    }

    const int nTiles = N / NT;
    const int S = nTiles * 2;             // 2 K-stages (256 each) per N tile

    // Stage s: nt = s>>1 (column tile), khalf = s&1 (K offset in {0,256}).
    auto stage_gptr = [&](int s) {
        return eh + (size_t)((s >> 1) * NT) * DD + (s & 1) * KH;
    };

#ifdef USE_TDM
    if (wave == 0) tdm_issue_tile(stage_gptr(0),
                                  (unsigned int)(uintptr_t)&sB[0][0][0]);
#endif

    v8f c = {};
    for (int s = 0; s < S; ++s) {
        const int buf = s & 1;

#ifdef USE_TDM
        if (wave == 0) __builtin_amdgcn_s_wait_tensorcnt(0);  // stage s landed
        __syncthreads();
        if (wave == 0 && s + 1 < S)
            tdm_issue_tile(stage_gptr(s + 1),
                           (unsigned int)(uintptr_t)&sB[buf ^ 1][0][0]);
#else
        // Fallback: cooperative synchronous copy of the 64x256 tile.
        {
            const _Float16* g = stage_gptr(s);
            for (int i = tid; i < NT * (KH / 8); i += 256) {
                const int r = i >> 5, o = (i & 31) * 8;
                *(v8h*)&sB[buf][r][o] = *(const v8h*)(g + (size_t)r * DD + o);
            }
            __syncthreads();
        }
#endif

        // Compute: B fragment column = this wave's n, from LDS.
        const _Float16* bcol = &sB[buf][ni * 16 + (lane & 15)][0];
        if ((s & 1) == 0) compute_stage<0>(c, aF, bcol, kSel);
        else              compute_stage<8>(c, aF, bcol, kSel);

        if (s & 1) {       // finished K=512 for N tile nt
            const int nt    = s >> 1;
            const int nBase = nt * NT;
            const int col   = ni * 16 + (lane & 15);
            const int rbase = mi * 16 + ((lane & 16) ? 8 : 0);
            #pragma unroll
            for (int r = 0; r < 8; ++r)
                sTile[rbase + r][col] = c[r];
            __syncthreads();

            if (tid < MT) {
                const int r = tid;
                float mv = sMin[r];
                int   mp = sMinPos[r];
                for (int j = 0; j < NT; ++j) {
                    const float sc = sTile[r][j];
                    if (sc > mv) {
                        sTopV[r][mp] = sc;
                        sTopI[r][mp] = nBase + j;
                        mv = sTopV[r][0]; mp = 0;
                        for (int q = 1; q < KK; ++q) {
                            const float v = sTopV[r][q];
                            if (v < mv) { mv = v; mp = q; }
                        }
                    }
                }
                sMin[r] = mv; sMinPos[r] = mp;
            }
            c = (v8f){};
        }
        // Selection vs. next tile-store is protected by the stage barrier(s).
#ifndef USE_TDM
        __syncthreads();
#endif
    }

    __syncthreads();

    // Final per-row descending sort + output.
    if (tid < MT) {
        const int r   = tid;
        const int row = mRowBlk + r;
        for (int a = 0; a < KK; ++a) {
            int best = a;
            float bv = sTopV[r][a];
            for (int b2 = a + 1; b2 < KK; ++b2) {
                const float v = sTopV[r][b2];
                if (v > bv) { bv = v; best = b2; }
            }
            const float tv = sTopV[r][a]; sTopV[r][a] = sTopV[r][best]; sTopV[r][best] = tv;
            const int   ti = sTopI[r][a]; sTopI[r][a] = sTopI[r][best]; sTopI[r][best] = ti;
            simsOut[(size_t)row * KK + a] = sTopV[r][a];
            idxOut [(size_t)row * KK + a] = sTopI[r][a];
        }
    }
}

// ---------------------------------------------------------------------------
extern "C" void kernel_launch(void* const* d_in, const int* in_sizes, int n_in,
                              void* d_out, int out_size, void* d_ws, size_t ws_size,
                              hipStream_t stream) {
    const float* z = (const float*)d_in[0];
    const float* e = (const float*)d_in[1];
    // d_in[2] holds k; fixed to 64 here.

    const int B = in_sizes[0] / DD;   // 4096
    const int N = in_sizes[1] / DD;   // 16384

    _Float16* qh = (_Float16*)d_ws;
    _Float16* eh = qh + (size_t)B * DD;

    float* simsOut = (float*)d_out;
    int*   idxOut  = (int*)((float*)d_out + (size_t)B * KK);

    const int rows = B + N;
    cr_normalize_f16<<<(rows + 7) / 8, 256, 0, stream>>>(z, e, qh, eh, B, N);

    cr_gemm_topk<<<B / MT, 256, 0, stream>>>(qh, eh, simsOut, idxOut, B, N);
}
